// HyperRadialNeuralFourierCelularAutomata_6219112645271
// MI455X (gfx1250) — matrix operand order repaired and checked by reading, not compiled, not run on hardware
//
#include <hip/hip_runtime.h>
#include <hip/hip_bf16.h>

// Sizes from the reference
#define BB   16
#define SS   7
#define PP   49
#define BITS 32
#define HDC  32
#define RBF  32

typedef _Float16 v16h __attribute__((ext_vector_type(16)));
typedef float    v8f  __attribute__((ext_vector_type(8)));

// ---------------------------------------------------------------------------
// Kernel 1: fold the 7-way XOR into packed channel-bit words.
// actbits[b][p][bit][h] : bit c of the u32 is rbf_dist[b,c,p,bit,h] in {0,1}.
// ---------------------------------------------------------------------------
__global__ void pack_act_kernel(const float* __restrict__ data,      // [B,4S,S]
                                const float* __restrict__ structure, // [B,S,S]
                                const int*   __restrict__ mi,        // [B,32]
                                const int*   __restrict__ mo,        // [B,32]
                                const int*   __restrict__ hdc,       // [B,P,32,32]
                                const float* __restrict__ rbf,       // [RBF,P,32,32]
                                unsigned*    __restrict__ actbits)   // [B,P,32,32]
{
    int idx = blockIdx.x * blockDim.x + threadIdx.x;   // one thread per (p,bit,h)
    if (idx >= PP * BITS * HDC) return;
    int h   = idx & 31;
    int bit = (idx >> 5) & 31;
    int p   = idx >> 10;

    // channel-packed (rbf >= 1.0) word, shared across batches
    unsigned rw = 0u;
    #pragma unroll
    for (int c = 0; c < RBF; ++c) {
        float v = rbf[((c * PP + p) * BITS + bit) * HDC + h];
        rw |= (v >= 1.0f ? 1u : 0u) << c;
    }

    for (int b = 0; b < BB; ++b) {
        unsigned xr = __float_as_uint(data[b * 196 +       p])   // r
                    ^ __float_as_uint(data[b * 196 +  49 + p])   // g
                    ^ __float_as_uint(data[b * 196 +  98 + p])   // b
                    ^ __float_as_uint(data[b * 196 + 147 + p])   // a
                    ^ __float_as_uint(structure[b * PP + p]);    // s
        unsigned mbit = ((xr >> bit) & 1u)
                      ^ ((unsigned)mi[b * BITS + bit] & 1u)
                      ^ ((unsigned)mo[b * BITS + bit] & 1u);
        unsigned hv = ((unsigned)hdc[((b * PP + p) * BITS + bit) * HDC + h] & 1u) ^ mbit;
        actbits[((b * PP + p) * BITS + bit) * HDC + h] = rw ^ (hv ? 0xFFFFFFFFu : 0u);
    }
}

// ---------------------------------------------------------------------------
// Kernel 2: pre-swizzle nca_w (f32 [32][32][3][3][3]) into f16 WMMA A-fragments.
// Layout: wsw[tap][mtile][lane][16 halfs] -> one 32B contiguous chunk per lane.
// A order used here (contiguous K, B matches): lanes 0-15 hold M=lane, K=0..15;
// lanes 16-31 hold M=lane-16, K=16..31; dword d holds a K pair.
// ---------------------------------------------------------------------------
__global__ void swizzle_w_kernel(const float* __restrict__ nw,
                                 _Float16*    __restrict__ wsw)
{
    int t = blockIdx.x * blockDim.x + threadIdx.x;
    if (t >= 27 * 2 * 32) return;
    int lane  = t & 31;
    int mtile = (t >> 5) & 1;
    int tap   = t >> 6;
    int dp = tap / 9, db = (tap / 3) % 3, dh = tap % 3;
    int oc = mtile * 16 + (lane & 15);
    int kb = (lane < 16) ? 0 : 16;

    _Float16* dst = wsw + t * 16;
    #pragma unroll
    for (int d = 0; d < 8; ++d) {
        int k0 = kb + 2 * d;
        #pragma unroll
        for (int e = 0; e < 2; ++e) {
            int ic = k0 + e;
            dst[2 * d + e] =
                (_Float16)nw[(((oc * RBF + ic) * 3 + dp) * 3 + db) * 3 + dh];
        }
    }
}

// ---------------------------------------------------------------------------
// Kernel 3: implicit-GEMM 3x3x3 conv via V_WMMA_F32_16X16X32_F16 + epilogue.
// One workgroup per (b, p, bitgroup-of-8). Padded f16 activation tile in LDS:
// tile[pp 3][bit 10][h 34][ic 32] = 65,280 B, channels innermost so each B
// fragment K-half is 32 contiguous bytes -> 2x ds_load_b128 per lane.
// Tap loop fully unrolled: all DS / global offsets become immediates.
// ---------------------------------------------------------------------------
__global__ __launch_bounds__(256, 1)
void conv_wmma_kernel(const unsigned* __restrict__ actbits,
                      const _Float16* __restrict__ wsw,
                      const float*    __restrict__ ncab,
                      float*          __restrict__ out)
{
    // flat, byte-addressed LDS tile: [3][10][34][32] f16
    __shared__ __align__(64) _Float16 tile[3 * 10 * 34 * 32];

    int blk = blockIdx.x;               // b*196 + p*4 + bg
    int bg  = blk & 3;
    int p   = (blk >> 2) % PP;
    int b   = blk / (PP * 4);
    int tid = threadIdx.x;
    int bitbase = bg * 8;

    // ---- Fill LDS: unpack packed channel bits (with halo, zero padding) ----
    for (int site = tid; site < 3 * 10 * 34; site += 256) {
        int sh = site % 34;
        int sb = (site / 34) % 10;
        int pp = site / 340;
        int gp   = p + pp - 1;
        int gbit = bitbase + sb - 1;
        int gh   = sh - 1;
        unsigned w = 0u;
        if (gp >= 0 && gp < PP && gbit >= 0 && gbit < BITS && gh >= 0 && gh < HDC)
            w = actbits[((b * PP + gp) * BITS + gbit) * HDC + gh];
        unsigned* dst = (unsigned*)(tile + (size_t)site * 32);
        #pragma unroll
        for (int j = 0; j < 16; ++j) {
            unsigned pair = (((w >> (2 * j)) & 1u) ? 0x3C00u : 0u)
                          | (((w >> (2 * j + 1)) & 1u) ? 0x3C000000u : 0u);
            dst[j] = pair;
        }
    }
    __syncthreads();

    int lane  = tid & 31;
    int wave  = tid >> 5;      // 8 waves: one local bit row each
    int bl    = wave;          // local bit index 0..7
    int col   = lane & 15;     // B/D column (h position within N-tile)
    int khalf = lane >> 4;     // B-matrix K half per lane

    // per-lane constant bases; everything below folds to immediate offsets
    const char* wbase = (const char*)wsw + lane * 32;                 // A frags
    const _Float16* tbase = tile + (bl * 34 + col) * 32 + khalf * 16; // B frags

    v8f acc[2][2] = {};        // [ntile(h-half)][mtile(oc-half)]

    #pragma unroll
    for (int dp = 0; dp < 3; ++dp) {
        #pragma unroll
        for (int db = 0; db < 3; ++db) {
            #pragma unroll
            for (int dh = 0; dh < 3; ++dh) {
                const int tap = (dp * 3 + db) * 3 + dh;
                // A fragments (weights), 32B/lane, base + immediate offset
                v16h a0 = *(const v16h*)(wbase + (tap * 2 + 0) * 1024);
                v16h a1 = *(const v16h*)(wbase + (tap * 2 + 1) * 1024);
                // B fragments from LDS, base + immediate ds offsets
                const _Float16* t0 =
                    tbase + ((dp * 10 + db) * 34 + dh) * 32;
                v16h b0 = *(const v16h*)(t0);
                v16h b1 = *(const v16h*)(t0 + 16 * 32);

                acc[0][0] = __builtin_amdgcn_wmma_f32_16x16x32_f16(
                    false, a0, false, b0, (short)0, acc[0][0], false, false);
                acc[0][1] = __builtin_amdgcn_wmma_f32_16x16x32_f16(
                    false, a1, false, b0, (short)0, acc[0][1], false, false);
                acc[1][0] = __builtin_amdgcn_wmma_f32_16x16x32_f16(
                    false, a0, false, b1, (short)0, acc[1][0], false, false);
                acc[1][1] = __builtin_amdgcn_wmma_f32_16x16x32_f16(
                    false, a1, false, b1, (short)0, acc[1][1], false, false);
            }
        }
    }

    // ---- Epilogue: out = x + tanh(dx + bias), coalesced along h ----
    int gbit = bitbase + bl;
    #pragma unroll
    for (int nt = 0; nt < 2; ++nt) {
        int h = nt * 16 + col;
        #pragma unroll
        for (int mt = 0; mt < 2; ++mt) {
            #pragma unroll
            for (int j = 0; j < 8; ++j) {
                int oc = mt * 16 + j + (lane >= 16 ? 8 : 0);  // D-matrix row map
                float x  = (float)tile[((10 + bl + 1) * 34 + h + 1) * 32 + oc];
                float dv = acc[nt][mt][j] + ncab[oc];
                out[(((b * RBF + oc) * PP + p) * BITS + gbit) * HDC + h]
                    = x + tanhf(dv);
            }
        }
    }
}

// ---------------------------------------------------------------------------
extern "C" void kernel_launch(void* const* d_in, const int* in_sizes, int n_in,
                              void* d_out, int out_size, void* d_ws, size_t ws_size,
                              hipStream_t stream) {
    const float* data      = (const float*)d_in[0];
    const float* structure = (const float*)d_in[1];
    const int*   mi        = (const int*)d_in[2];
    const int*   mo        = (const int*)d_in[3];
    const int*   hdc       = (const int*)d_in[4];
    const float* rbf       = (const float*)d_in[5];
    const float* nw        = (const float*)d_in[6];
    const float* nb        = (const float*)d_in[7];
    float* out = (float*)d_out;

    // workspace layout: packed activations (3.2 MB), then swizzled f16 weights
    unsigned* actbits = (unsigned*)d_ws;
    _Float16* wsw = (_Float16*)((char*)d_ws + (size_t)BB * PP * BITS * HDC * 4);

    int nwords = PP * BITS * HDC;                       // 50176
    pack_act_kernel<<<(nwords + 255) / 256, 256, 0, stream>>>(
        data, structure, mi, mo, hdc, rbf, actbits);

    swizzle_w_kernel<<<(27 * 2 * 32 + 255) / 256, 256, 0, stream>>>(nw, wsw);

    conv_wmma_kernel<<<BB * PP * 4, 256, 0, stream>>>(actbits, wsw, nb, out);
}
